// CrossAttentionModule_64811056497194
// MI455X (gfx1250) — matrix operand order repaired and checked by reading, not compiled, not run on hardware
//
#include <hip/hip_runtime.h>

// ---------------------------------------------------------------------------
// CDNA5 (gfx1250) cross-attention module, v3.1.
// All GEMMs run on v_wmma_f32_16x16x32_bf16 with split-bf16 operands
// (x ~= hi + lo, products hi*hi + hi*lo + lo*hi, f32 accumulation) giving
// near-fp32 accuracy at bf16 matrix-pipe rate.  Operands are pre-split into
// separate hi/lo ushort planes once; the GEMM hot loop is double-buffered
// async global->LDS tile copies (ASYNCcnt) overlapped with WMMA issue.
// ---------------------------------------------------------------------------

typedef unsigned short u16;
typedef __attribute__((ext_vector_type(16))) __bf16 v16bf;
typedef __attribute__((ext_vector_type(8)))  float  v8f;

struct U16x16 { uint4 a, b; };

// -------- async global->LDS (gfx1250) with portable fallback ---------------
#if defined(__has_builtin)
# if __has_builtin(__builtin_amdgcn_global_load_async_to_lds_b128) && \
     __has_builtin(__builtin_amdgcn_s_wait_asynccnt)
#  define USE_ASYNC_LDS 1
# endif
#endif
#ifndef USE_ASYNC_LDS
# define USE_ASYNC_LDS 0
#endif

#if USE_ASYNC_LDS
typedef int v4i __attribute__((vector_size(16)));
typedef __attribute__((address_space(1))) v4i gv4i;   // global (__device__) int4
typedef __attribute__((address_space(3))) v4i sv4i;   // LDS (__shared__) int4
__device__ __forceinline__ void ld_async_b128(const u16* g, u16* l) {
    __builtin_amdgcn_global_load_async_to_lds_b128((gv4i*)g, (sv4i*)l, 0, 0);
}
__device__ __forceinline__ void async_wait() {
    __builtin_amdgcn_s_wait_asynccnt(0);
}
#else
__device__ __forceinline__ void ld_async_b128(const u16* g, u16* l) {
    *(uint4*)l = *(const uint4*)g;
}
__device__ __forceinline__ void async_wait() {}
#endif

__device__ __forceinline__ u16 f2bf(float f) {
    unsigned u = __float_as_uint(f);
    u += 0x7FFFu + ((u >> 16) & 1u);          // round-to-nearest-even
    return (u16)(u >> 16);
}
__device__ __forceinline__ float bf2f(u16 s) {
    return __uint_as_float((unsigned)s << 16);
}
__device__ __forceinline__ v16bf frag16(const uint4 a, const uint4 b) {
    U16x16 t{a, b};
    return __builtin_bit_cast(v16bf, t);
}

// ---------------------------------------------------------------------------
// Batched GEMM on pre-split bf16 planes:
//   Y[b] = scale * opA(A[b]) * opB(B[b]) (+ bias)
//   opA(A)(m,k) = TA ? A[k*lda+m] : A[m*lda+k]
//   opB(B)(k,n) = TB ? B[n*ldb+k] : B[k*ldb+n]
// Block: 256 threads / 8 waves; block tile 64(M) x 128(N); K-step 32.
// Wave tile 32x32 -> 2x2 accumulators, 12 WMMA per wave per K-step.
// Double-buffered LDS tiles: tile t+1 is staged (async where contiguous)
// while tile t feeds the WMMAs.  Requires 64|M, 128|N, 32|K.
// ---------------------------------------------------------------------------
template <bool TA, bool TB, bool HAS_BIAS, bool SPLIT_OUT>
__global__ __launch_bounds__(256)
void gemm_wmma(const u16* __restrict__ Ahg, const u16* __restrict__ Alg,
               const u16* __restrict__ Bhg, const u16* __restrict__ Blg,
               const float* __restrict__ bias,
               float* __restrict__ Yf, u16* __restrict__ Yh, u16* __restrict__ Yl,
               int K, int lda, int ldb, int ldy, float scale,
               long long sA, long long sB, long long sY)
{
    // A tiles m-major (k contiguous), B tiles n-major (k contiguous) so each
    // lane's 16 fragment K-values are contiguous ds_load_b128 pairs.
    __shared__ u16 Ah[2][64][40];
    __shared__ u16 Al[2][64][40];
    __shared__ u16 Bh[2][128][40];
    __shared__ u16 Bl[2][128][40];

    const int tid  = threadIdx.x;
    const int lane = tid & 31;
    const int wave = tid >> 5;
    const int wrow = (wave >> 2) * 32;   // 0 / 32
    const int wcol = (wave & 3) * 32;    // 0 / 32 / 64 / 96
    const int m0   = blockIdx.y * 64;
    const int n0   = blockIdx.x * 128;

    Ahg += blockIdx.z * sA;  Alg += blockIdx.z * sA;
    Bhg += blockIdx.z * sB;  Blg += blockIdx.z * sB;
    if (SPLIT_OUT) { Yh += blockIdx.z * sY; Yl += blockIdx.z * sY; }
    else           { Yf += blockIdx.z * sY; }

    v8f acc[2][2] = {};

    // wave32 WMMA fragment geometry (CDNA5 ISA §7.12.2)
    const int fr  = lane & 15;           // A/B: M or N index; D: column
    const int akh = (lane >> 4) * 8;     // A: K half select
    const int bkb = (lane >> 4) * 8;     // B: K half select

    // ---- stage one 64x32 (A) + 32x128 (B) tile pair into buffer `buf` ----
    auto stage = [&](int buf, int k0) {
        if (!TA) {
            const int r  = tid >> 2;              // 0..63 (m)
            const int kc = (tid & 3) * 8;         // 0,8,16,24 (k)
            const size_t g = (size_t)(m0 + r) * lda + (k0 + kc);
            ld_async_b128(Ahg + g, &Ah[buf][r][kc]);
            ld_async_b128(Alg + g, &Al[buf][r][kc]);
        } else {
            const int kk = tid >> 3;              // 0..31 (k)
            const int mc = (tid & 7) * 8;         // 0..56 (m)
            const size_t g = (size_t)(k0 + kk) * lda + (m0 + mc);
            const uint4 vh = *(const uint4*)(Ahg + g);
            const uint4 vl = *(const uint4*)(Alg + g);
            const unsigned wh[4] = {vh.x, vh.y, vh.z, vh.w};
            const unsigned wl[4] = {vl.x, vl.y, vl.z, vl.w};
            #pragma unroll
            for (int j = 0; j < 4; ++j) {
                Ah[buf][mc + 2*j    ][kk] = (u16)wh[j];
                Ah[buf][mc + 2*j + 1][kk] = (u16)(wh[j] >> 16);
                Al[buf][mc + 2*j    ][kk] = (u16)wl[j];
                Al[buf][mc + 2*j + 1][kk] = (u16)(wl[j] >> 16);
            }
        }
        if (!TB) {
            const int kk = tid >> 4;              // 0..15 (k; also k+16)
            const int nc = (tid & 15) * 8;        // 0..120 (n)
            #pragma unroll
            for (int s = 0; s < 2; ++s) {
                const int k2 = kk + s * 16;
                const size_t g = (size_t)(k0 + k2) * ldb + (n0 + nc);
                const uint4 vh = *(const uint4*)(Bhg + g);
                const uint4 vl = *(const uint4*)(Blg + g);
                const unsigned wh[4] = {vh.x, vh.y, vh.z, vh.w};
                const unsigned wl[4] = {vl.x, vl.y, vl.z, vl.w};
                #pragma unroll
                for (int j = 0; j < 4; ++j) {
                    Bh[buf][nc + 2*j    ][k2] = (u16)wh[j];
                    Bh[buf][nc + 2*j + 1][k2] = (u16)(wh[j] >> 16);
                    Bl[buf][nc + 2*j    ][k2] = (u16)wl[j];
                    Bl[buf][nc + 2*j + 1][k2] = (u16)(wl[j] >> 16);
                }
            }
        } else {
            const int nn = tid >> 2;              // 0..63 (n; also n+64)
            const int kc = (tid & 3) * 8;         // 0,8,16,24 (k)
            #pragma unroll
            for (int s = 0; s < 2; ++s) {
                const size_t g = (size_t)(n0 + nn + 64*s) * ldb + (k0 + kc);
                ld_async_b128(Bhg + g, &Bh[buf][nn + 64*s][kc]);
                ld_async_b128(Blg + g, &Bl[buf][nn + 64*s][kc]);
            }
        }
    };

    const int nk = K >> 5;
    stage(0, 0);
    async_wait();
    __syncthreads();

    for (int t = 0; t < nk; ++t) {
        const int buf = t & 1;
        if (t + 1 < nk) stage(buf ^ 1, (t + 1) << 5);   // overlap next tile copy

        // -------- fragments: contiguous 16B LDS reads -----------------------
        v16bf afh[2], afl[2], bfh[2], bfl[2];
        #pragma unroll
        for (int mi = 0; mi < 2; ++mi) {
            const u16* ph = &Ah[buf][wrow + mi*16 + fr][0];
            const u16* pl = &Al[buf][wrow + mi*16 + fr][0];
            afh[mi] = frag16(*(const uint4*)(ph + akh), *(const uint4*)(ph + akh + 16));
            afl[mi] = frag16(*(const uint4*)(pl + akh), *(const uint4*)(pl + akh + 16));
        }
        #pragma unroll
        for (int ni = 0; ni < 2; ++ni) {
            const u16* ph = &Bh[buf][wcol + ni*16 + fr][0];
            const u16* pl = &Bl[buf][wcol + ni*16 + fr][0];
            bfh[ni] = frag16(*(const uint4*)(ph + bkb), *(const uint4*)(ph + bkb + 16));
            bfl[ni] = frag16(*(const uint4*)(pl + bkb), *(const uint4*)(pl + bkb + 16));
        }

        // -------- 12 WMMAs: hi*hi + hi*lo + lo*hi ---------------------------
        #pragma unroll
        for (int mi = 0; mi < 2; ++mi)
        #pragma unroll
        for (int ni = 0; ni < 2; ++ni) {
            acc[mi][ni] = __builtin_amdgcn_wmma_f32_16x16x32_bf16(
                false, afh[mi], false, bfh[ni], (short)0, acc[mi][ni], false, false);
            acc[mi][ni] = __builtin_amdgcn_wmma_f32_16x16x32_bf16(
                false, afh[mi], false, bfl[ni], (short)0, acc[mi][ni], false, false);
            acc[mi][ni] = __builtin_amdgcn_wmma_f32_16x16x32_bf16(
                false, afl[mi], false, bfh[ni], (short)0, acc[mi][ni], false, false);
        }

        async_wait();          // next tile's async copies landed in LDS
        __syncthreads();       // visible to all waves before buffer flip
    }

    // -------- epilogue: D layout VGPR r -> M = r (+8 for upper lanes) -------
    const int rOff = (lane >> 4) * 8;
    #pragma unroll
    for (int mi = 0; mi < 2; ++mi) {
        const int gmB = m0 + wrow + mi*16 + rOff;
        float bv[8];
        if (HAS_BIAS) {
            #pragma unroll
            for (int r = 0; r < 8; ++r) bv[r] = bias[gmB + r];
        }
        #pragma unroll
        for (int ni = 0; ni < 2; ++ni) {
            const int gn = n0 + wcol + ni*16 + fr;
            const size_t base = (size_t)gmB * ldy + gn;
            #pragma unroll
            for (int r = 0; r < 8; ++r) {
                float v = acc[mi][ni][r] * scale;
                if (HAS_BIAS) v += bv[r];
                const size_t idx = base + (size_t)r * ldy;
                if (SPLIT_OUT) {
                    const u16 h = f2bf(v);
                    Yh[idx] = h;
                    Yl[idx] = f2bf(v - bf2f(h));
                } else {
                    Yf[idx] = v;
                }
            }
        }
    }
}

// ---------------------------------------------------------------------------
// One-time fp32 -> (bf16 hi, bf16 lo) plane split.
// ---------------------------------------------------------------------------
__global__ __launch_bounds__(256)
void split_planes(const float* __restrict__ X, u16* __restrict__ H,
                  u16* __restrict__ L, long long n)
{
    const long long i = ((long long)blockIdx.x * 256 + threadIdx.x) * 4;
    if (i >= n) return;
    const float4 v = *(const float4*)(X + i);
    const float f[4] = {v.x, v.y, v.z, v.w};
    unsigned hq[2] = {0u, 0u}, lq[2] = {0u, 0u};
    #pragma unroll
    for (int j = 0; j < 4; ++j) {
        const u16 h = f2bf(f[j]);
        const u16 l = f2bf(f[j] - bf2f(h));
        hq[j >> 1] |= (unsigned)h << ((j & 1) * 16);
        lq[j >> 1] |= (unsigned)l << ((j & 1) * 16);
    }
    *(uint2*)(H + i) = make_uint2(hq[0], hq[1]);
    *(uint2*)(L + i) = make_uint2(lq[0], lq[1]);
}

// ---------------------------------------------------------------------------
// Row softmax over M columns (in-place exp in S), emitting split bf16 planes.
// ---------------------------------------------------------------------------
__global__ __launch_bounds__(256)
void softmax_split(float* __restrict__ S, u16* __restrict__ H,
                   u16* __restrict__ L, int M)
{
    const size_t row = blockIdx.x;
    float* p = S + row * M;
    __shared__ float red[256];
    const int tid = threadIdx.x;

    float m = -INFINITY;
    for (int j = tid; j < M; j += 256) m = fmaxf(m, p[j]);
    red[tid] = m; __syncthreads();
    for (int s = 128; s > 0; s >>= 1) {
        if (tid < s) red[tid] = fmaxf(red[tid], red[tid + s]);
        __syncthreads();
    }
    const float rowmax = red[0]; __syncthreads();

    float sum = 0.f;
    for (int j = tid; j < M; j += 256) {
        const float e = expf(p[j] - rowmax);
        p[j] = e;
        sum += e;
    }
    red[tid] = sum; __syncthreads();
    for (int s = 128; s > 0; s >>= 1) {
        if (tid < s) red[tid] += red[tid + s];
        __syncthreads();
    }
    const float inv = 1.f / red[0];
    for (int j = tid; j < M; j += 256) {
        const float v = p[j] * inv;
        const u16 h = f2bf(v);
        H[row * M + j] = h;
        L[row * M + j] = f2bf(v - bf2f(h));
    }
}

// ---------------------------------------------------------------------------
// F_final = att[:,0] + att[:,1];  passthrough copies of F_rgb / F_indices.
// ---------------------------------------------------------------------------
__global__ __launch_bounds__(256)
void finalize_outputs(const float* __restrict__ Frgb, const float* __restrict__ Find,
                      const float* __restrict__ attn, float* __restrict__ out,
                      long long CN, long long FCN)
{
    const long long i = ((long long)blockIdx.x * blockDim.x + threadIdx.x) * 4;
    if (i >= FCN) return;
    const long long b = i / CN, r = i % CN;
    const float4 a0 = *(const float4*)(attn + b * 2 * CN + r);
    const float4 a1 = *(const float4*)(attn + b * 2 * CN + CN + r);
    float4 f;
    f.x = a0.x + a1.x; f.y = a0.y + a1.y; f.z = a0.z + a1.z; f.w = a0.w + a1.w;
    *(float4*)(out + i)           = f;
    *(float4*)(out + FCN + i)     = *(const float4*)(Frgb + i);
    *(float4*)(out + 2 * FCN + i) = *(const float4*)(Find + i);
}

// ---------------------------------------------------------------------------
extern "C" void kernel_launch(void* const* d_in, const int* in_sizes, int n_in,
                              void* d_out, int out_size, void* d_ws, size_t ws_size,
                              hipStream_t stream)
{
    (void)in_sizes; (void)n_in; (void)out_size; (void)ws_size;

    const int C = 2048, NP = 1024, BT = 4;
    const long long CN  = (long long)C * NP;     // 2,097,152  (per-batch C*H*W)
    const long long FCN = (long long)BT * CN;    // 8,388,608
    const long long CC  = (long long)C * C;      // 4,194,304
    const long long NN  = (long long)NP * NP;    // per-batch scores
    const long long SNN = (long long)BT * NN;    // 4,194,304

    const float* Frgb    = (const float*)d_in[0];
    const float* Find    = (const float*)d_in[1];
    const float* W_q_rgb = (const float*)d_in[2];  const float* b_q_rgb = (const float*)d_in[3];
    const float* W_k_rgb = (const float*)d_in[4];  const float* b_k_rgb = (const float*)d_in[5];
    const float* W_v_rgb = (const float*)d_in[6];  const float* b_v_rgb = (const float*)d_in[7];
    const float* W_q_ind = (const float*)d_in[8];  const float* b_q_ind = (const float*)d_in[9];
    const float* W_k_ind = (const float*)d_in[10]; const float* b_k_ind = (const float*)d_in[11];
    const float* W_v_ind = (const float*)d_in[12]; const float* b_v_ind = (const float*)d_in[13];

    // ---------------- workspace carve-up (~252 MB) ----------------
    u16* w = (u16*)d_ws;
    u16* Xrgb_h = w;  w += FCN;   u16* Xrgb_l = w;  w += FCN;
    u16* Xind_h = w;  w += FCN;   u16* Xind_l = w;  w += FCN;
    u16* Wsp    = w;  w += 6 * CC;   // per pass: Wq_h, Wq_l, Wk_h, Wk_l, Wv_h, Wv_l
    u16* Qh = w; w += FCN;  u16* Ql = w; w += FCN;
    u16* Kh = w; w += FCN;  u16* Kl = w; w += FCN;
    u16* Vh = w; w += FCN;  u16* Vl = w; w += FCN;
    u16* At_h = w; w += SNN; u16* At_l = w; w += SNN;
    float* Sc = (float*)w;           // SNN floats

    float* out     = (float*)d_out;
    float* attnOut = out + 3 * FCN;  // attention_weights region [B,2,C,H,W]

    const dim3 blk(256);
    const dim3 gConv(NP / 128, C / 64, BT);    // (8, 32, 4)
    const dim3 gScore(NP / 128, NP / 64, BT);  // (8, 16, 4)
    const float invSqrtC = 0.022097086912079608f;  // 1/sqrt(2048)

    // one-time split of the two feature maps
    split_planes<<<(unsigned)(FCN / 4 / 256), blk, 0, stream>>>(Frgb, Xrgb_h, Xrgb_l, FCN);
    split_planes<<<(unsigned)(FCN / 4 / 256), blk, 0, stream>>>(Find, Xind_h, Xind_l, FCN);

    for (int pass = 0; pass < 2; ++pass) {
        // pass 0: Q from RGB, K/V from indices.  pass 1: swapped.
        const u16*   Xq_h = (pass == 0) ? Xrgb_h : Xind_h;
        const u16*   Xq_l = (pass == 0) ? Xrgb_l : Xind_l;
        const u16*   Xk_h = (pass == 0) ? Xind_h : Xrgb_h;
        const u16*   Xk_l = (pass == 0) ? Xind_l : Xrgb_l;
        const float* Wq   = (pass == 0) ? W_q_rgb : W_q_ind;
        const float* bq   = (pass == 0) ? b_q_rgb : b_q_ind;
        const float* Wk   = (pass == 0) ? W_k_ind : W_k_rgb;
        const float* bk   = (pass == 0) ? b_k_ind : b_k_rgb;
        const float* Wv   = (pass == 0) ? W_v_ind : W_v_rgb;
        const float* bv   = (pass == 0) ? b_v_ind : b_v_rgb;
        float* dst = attnOut + pass * CN;    // slot in [B,2,C,N]

        // split this pass's weights (hi/lo planes, reused region)
        split_planes<<<(unsigned)(CC / 4 / 256), blk, 0, stream>>>(Wq, Wsp + 0 * CC, Wsp + 1 * CC, CC);
        split_planes<<<(unsigned)(CC / 4 / 256), blk, 0, stream>>>(Wk, Wsp + 2 * CC, Wsp + 3 * CC, CC);
        split_planes<<<(unsigned)(CC / 4 / 256), blk, 0, stream>>>(Wv, Wsp + 4 * CC, Wsp + 5 * CC, CC);

        // 1x1 convs: [C,C] x [C,NP] + bias -> split planes  (M=C, N=NP, K=C)
        gemm_wmma<false, false, true, true><<<gConv, blk, 0, stream>>>(
            Wsp + 0 * CC, Wsp + 1 * CC, Xq_h, Xq_l, bq, nullptr, Qh, Ql,
            C, C, NP, NP, 1.f, 0, CN, CN);
        gemm_wmma<false, false, true, true><<<gConv, blk, 0, stream>>>(
            Wsp + 2 * CC, Wsp + 3 * CC, Xk_h, Xk_l, bk, nullptr, Kh, Kl,
            C, C, NP, NP, 1.f, 0, CN, CN);
        gemm_wmma<false, false, true, true><<<gConv, blk, 0, stream>>>(
            Wsp + 4 * CC, Wsp + 5 * CC, Xk_h, Xk_l, bv, nullptr, Vh, Vl,
            C, C, NP, NP, 1.f, 0, CN, CN);

        // scores = (Q^T K) / sqrt(C)  (M=NP, N=NP, K=C, A transposed) -> f32
        gemm_wmma<true, false, false, false><<<gScore, blk, 0, stream>>>(
            Qh, Ql, Kh, Kl, nullptr, Sc, nullptr, nullptr,
            C, NP, NP, NP, invSqrtC, CN, CN, NN);

        // softmax rows -> split attn planes
        softmax_split<<<BT * NP, blk, 0, stream>>>(Sc, At_h, At_l, NP);

        // out = V * attn^T  (M=C, N=NP, K=NP, B transposed) -> f32 into d_out
        gemm_wmma<false, true, false, false><<<gConv, blk, 0, stream>>>(
            Vh, Vl, At_h, At_l, nullptr, dst, nullptr, nullptr,
            NP, NP, NP, NP, 1.f, CN, NN, 2 * CN);
    }

    finalize_outputs<<<(unsigned)((FCN / 4 + 255) / 256), blk, 0, stream>>>(
        Frgb, Find, attnOut, out, CN, FCN);
}
